// SelfAttn_88931592831715
// MI455X (gfx1250) — matrix operand order repaired and checked by reading, not compiled, not run on hardware
//
#include <hip/hip_runtime.h>
#include <hip/hip_bf16.h>

typedef __bf16 bf16;
typedef __attribute__((ext_vector_type(16))) __bf16 v16bf;
typedef __attribute__((ext_vector_type(8)))  float  v8f;

static constexpr int kB  = 8;
static constexpr int kC  = 256;
static constexpr int kH  = 64;
static constexpr int kW  = 64;
static constexpr int kN  = kH * kW;      // 4096
static constexpr int kM  = kN / 4;       // 1024 pooled positions
static constexpr int kCq = kC / 8;       // 32 (theta/phi channels)
static constexpr int kCg = kC / 2;       // 128 (g channels)
static constexpr int kCcat = kCq + kCq + kCg; // 192
static constexpr int kCpg  = kCcat - kCq;     // 160 (phi+g full-res rows)

// ---------------------------------------------------------------------------
// WMMA helpers: D(16x16,f32) = A(16x32,bf16) * B(32x16,bf16) + C
// ---------------------------------------------------------------------------
__device__ __forceinline__ v8f wmma_bf16(v16bf a, v16bf b, v8f c) {
  return __builtin_amdgcn_wmma_f32_16x16x32_bf16(
      /*neg_a=*/false, a, /*neg_b=*/false, b,
      /*c_mod=*/(short)0, c, /*reuse_a=*/false, /*reuse_b=*/false);
}

// A-matrix 16x32 (MxK) gather from row-major bf16 [rows][ld].
// ISA layout: lanes 0-15 M=lane, K pairs {0..7,16..23}; lanes 16-31 K pairs {8..15,24..31}.
__device__ __forceinline__ v16bf load_a_16x32(const bf16* p, int ld, int lane) {
  const int m  = lane & 15;
  const int kb = (lane >> 4) << 3; // 0 or 8
  const bf16* row = p + (size_t)m * ld;
  v16bf a;
#pragma unroll
  for (int v = 0; v < 8; ++v) {
    const int k = (v < 4) ? (kb + 2 * v) : (16 + kb + 2 * (v - 4));
    a[2 * v + 0] = row[k + 0];
    a[2 * v + 1] = row[k + 1];
  }
  return a;
}

// B-matrix 32x16 (KxN) from row-major bf16 [K][ld]: N=lane%16, K {0..7,16..23} / {8..15,24..31} per lane half.
__device__ __forceinline__ v16bf load_b_32x16(const bf16* p, int ld, int lane) {
  const int n  = lane & 15;
  const int kb = (lane >> 4) << 3;
  v16bf b;
#pragma unroll
  for (int i = 0; i < 16; ++i) {
    const int k = (i < 8) ? (kb + i) : (16 + kb + (i - 8));
    b[i] = p[(size_t)k * ld + n];
  }
  return b;
}

// ---------------------------------------------------------------------------
// Kernel 0: bf16 weight prep. Wcat[192][256] = {w_theta; w_phi; w_g},
// WoT[c][oc] = w_o[oc][c] (so it can be a WMMA B operand directly).
// ---------------------------------------------------------------------------
__global__ void prep_weights(const float* __restrict__ wt, const float* __restrict__ wp,
                             const float* __restrict__ wg, const float* __restrict__ wo,
                             bf16* __restrict__ Wcat, bf16* __restrict__ WoT) {
  const int stride = gridDim.x * blockDim.x;
  for (int i = blockIdx.x * blockDim.x + threadIdx.x; i < kCcat * kC; i += stride) {
    const int r = i / kC, c = i % kC;
    float v = (r < kCq) ? wt[r * kC + c]
            : (r < 2 * kCq) ? wp[(r - kCq) * kC + c]
                            : wg[(r - 2 * kCq) * kC + c];
    Wcat[i] = (bf16)v;
  }
  for (int i = blockIdx.x * blockDim.x + threadIdx.x; i < kCg * kC; i += stride) {
    const int c = i / kC, oc = i % kC;
    WoT[i] = (bf16)wo[oc * kCg + c];
  }
}

// ---------------------------------------------------------------------------
// Kernel 1: fused projection GEMM. P[o,n] = sum_c Wcat[o,c]*(x[b,c,n]+pos[c]).
// One 16x16 D-tile per wave, K=256 in 8 WMMA steps.
// o<32  -> theta[b][o][n]  (bf16, row-major [32][4096])
// o>=32 -> projfull[b][o-32][n] (phi/g pre-pool, [160][4096])
// ---------------------------------------------------------------------------
__global__ void proj_kernel(const float* __restrict__ x, const float* __restrict__ pos,
                            const bf16* __restrict__ Wcat,
                            bf16* __restrict__ theta, bf16* __restrict__ projfull) {
  const int lane = threadIdx.x & 31;
  const int tile = blockIdx.x * (blockDim.x >> 5) + (threadIdx.x >> 5);
  const int NT = kN / 16;          // 256
  const int OT = kCcat / 16;       // 12
  const int b  = tile / (OT * NT);
  const int ot = (tile / NT) % OT;
  const int n0 = (tile % NT) * 16;
  const int o0 = ot * 16;

  v8f acc = {0.f, 0.f, 0.f, 0.f, 0.f, 0.f, 0.f, 0.f};
#pragma unroll
  for (int k0 = 0; k0 < kC; k0 += 32) {
    const v16bf A = load_a_16x32(Wcat + (size_t)o0 * kC + k0, kC, lane);
    const int kb = (lane >> 4) << 3;
    const int ncol = n0 + (lane & 15);
    if (k0 + 32 < kC)
      __builtin_prefetch(x + (size_t)(b * kC + k0 + 32 + kb) * kN + ncol, 0, 1);
    v16bf Bt;
#pragma unroll
    for (int s = 0; s < 16; ++s) {
      const int c = k0 + ((s < 8) ? (kb + s) : (16 + kb + (s - 8)));
      Bt[s] = (bf16)(x[(size_t)(b * kC + c) * kN + ncol] + pos[c]);
    }
    acc = wmma_bf16(A, Bt, acc);
  }

  const int nn = n0 + (lane & 15);
  const int hi = lane >> 4;
#pragma unroll
  for (int r = 0; r < 8; ++r) {
    const int o = o0 + r + 8 * hi;
    const bf16 v = (bf16)acc[r];
    if (o < kCq) theta[((size_t)b * kCq + o) * kN + nn] = v;
    else         projfull[((size_t)b * kCpg + (o - kCq)) * kN + nn] = v;
  }
}

// ---------------------------------------------------------------------------
// Kernel 2: 2x2 maxpool. Emits phiT[b][m][32] (A-operand layout) and
// gP[b][128][m] (row-major over m for A gathers in attention).
// ---------------------------------------------------------------------------
__global__ void pool_kernel(const bf16* __restrict__ projfull,
                            bf16* __restrict__ phiT, bf16* __restrict__ gP) {
  const int idx = blockIdx.x * blockDim.x + threadIdx.x;
  if (idx >= kB * kCpg * kM) return;
  const int m = idx % kM;
  const int c = (idx / kM) % kCpg;
  const int b = idx / (kM * kCpg);
  const int h2 = m / (kW / 2), w2 = m % (kW / 2);
  const size_t base = ((size_t)b * kCpg + c) * kN + (size_t)(2 * h2) * kW + 2 * w2;
  const float a0 = (float)projfull[base],      a1 = (float)projfull[base + 1];
  const float a2 = (float)projfull[base + kW], a3 = (float)projfull[base + kW + 1];
  const float mx = fmaxf(fmaxf(a0, a1), fmaxf(a2, a3));
  if (c < kCq) phiT[((size_t)b * kM + m) * kCq + c] = (bf16)mx;
  else         gP[((size_t)b * kCg + (c - kCq)) * kM + m] = (bf16)mx;
}

// ---------------------------------------------------------------------------
// Kernel 3: flash attention + PV in one pass. One wave owns 16 query rows.
// S[m,n] tiles => softmax reduction over M (in-lane + shfl_xor(16)).
// P staged to LDS (bf16 [32][16]) to become the B operand of acc += g * P.
// attn never hits global memory (saves 134 MB of HBM traffic).
// ---------------------------------------------------------------------------
__global__ void attn_kernel(const bf16* __restrict__ theta, const bf16* __restrict__ phiT,
                            const bf16* __restrict__ gP, bf16* __restrict__ attn_gT) {
  __shared__ bf16 Plds[4][32][16];
  const int lane = threadIdx.x & 31;
  const int wv   = threadIdx.x >> 5;
  const int tile = blockIdx.x * (blockDim.x >> 5) + wv;
  const int b  = tile / (kN / 16);
  const int n0 = (tile % (kN / 16)) * 16;
  const int hi = lane >> 4;

  const bf16* th = theta + (size_t)b * kCq * kN;  // [32][4096]
  const bf16* ph = phiT  + (size_t)b * kM * kCq;  // [1024][32]
  const bf16* gb = gP    + (size_t)b * kCg * kM;  // [128][1024]

  const v16bf Bth = load_b_32x16(th + n0, kN, lane);  // B: K=c, N=n (whole K=32)

  v8f acc[8];
#pragma unroll
  for (int t = 0; t < 8; ++t) acc[t] = v8f{0.f, 0.f, 0.f, 0.f, 0.f, 0.f, 0.f, 0.f};
  float rmax = -3.0e38f, rsum = 0.f;

  for (int m0 = 0; m0 < kM; m0 += 32) {
    const v8f z = {0.f, 0.f, 0.f, 0.f, 0.f, 0.f, 0.f, 0.f};
    v8f S0 = wmma_bf16(load_a_16x32(ph + (size_t)m0 * kCq, kCq, lane), Bth, z);
    v8f S1 = wmma_bf16(load_a_16x32(ph + (size_t)(m0 + 16) * kCq, kCq, lane), Bth, z);

    float tmax = -3.0e38f;
#pragma unroll
    for (int r = 0; r < 8; ++r) tmax = fmaxf(tmax, fmaxf(S0[r], S1[r]));
    tmax = fmaxf(tmax, __shfl_xor(tmax, 16, 32));   // lane pair shares column n
    const float nm = fmaxf(rmax, tmax);
    const float corr = __expf(rmax - nm);
    rmax = nm;
    rsum *= corr;
#pragma unroll
    for (int t = 0; t < 8; ++t) acc[t] *= corr;

    float ps = 0.f;
#pragma unroll
    for (int r = 0; r < 8; ++r) {
      const float p0 = __expf(S0[r] - rmax);
      const float p1 = __expf(S1[r] - rmax);
      ps += p0 + p1;
      Plds[wv][8 * hi + r][lane & 15]      = (bf16)p0;  // D layout -> [m][n]
      Plds[wv][16 + 8 * hi + r][lane & 15] = (bf16)p1;
    }
    rsum += ps;

    const v16bf Bp = load_b_32x16(&Plds[wv][0][0], 16, lane); // K=m (lane), N=n
#pragma unroll
    for (int t = 0; t < 8; ++t) {
      const v16bf Ag = load_a_16x32(gb + (size_t)(t * 16) * kM + m0, kM, lane);
      acc[t] = wmma_bf16(Ag, Bp, acc[t]);   // acc[c, n] over K=m
    }
  }

  rsum += __shfl_xor(rsum, 16, 32);
  const float inv = 1.0f / rsum;
  // Store attn_g transposed: attn_gT[b][n][c], so it is an A operand downstream.
  bf16* dst = attn_gT + ((size_t)b * kN + n0 + (lane & 15)) * kCg;
#pragma unroll
  for (int t = 0; t < 8; ++t)
#pragma unroll
    for (int r = 0; r < 8; ++r)
      dst[t * 16 + 8 * hi + r] = (bf16)(acc[t][r] * inv);
}

// ---------------------------------------------------------------------------
// Kernel 4: o = w_o @ attn_g fused with residual. D[n,oc] tiles, K=128.
// A = attn_gT rows, B = WoT. Lane writes 8 contiguous floats of out.
// ---------------------------------------------------------------------------
__global__ void out_kernel(const float* __restrict__ x, const bf16* __restrict__ WoT,
                           const bf16* __restrict__ attn_gT, const float* __restrict__ gamma,
                           float* __restrict__ out) {
  const int lane = threadIdx.x & 31;
  const int tile = blockIdx.x * (blockDim.x >> 5) + (threadIdx.x >> 5);
  const int NT  = kN / 16;   // 256
  const int OCT = kC / 16;   // 16
  const int b   = tile / (OCT * NT);
  const int ot  = (tile / NT) % OCT;
  const int n0  = (tile % NT) * 16;
  const int oc0 = ot * 16;

  const bf16* ag = attn_gT + (size_t)b * kN * kCg;

  v8f acc = {0.f, 0.f, 0.f, 0.f, 0.f, 0.f, 0.f, 0.f};
#pragma unroll
  for (int k0 = 0; k0 < kCg; k0 += 32) {
    const v16bf A  = load_a_16x32(ag + (size_t)n0 * kCg + k0, kCg, lane);
    const v16bf Bw = load_b_32x16(WoT + (size_t)k0 * kC + oc0, kC, lane);
    acc = wmma_bf16(A, Bw, acc);
  }

  const float gm = gamma[0];
  const int oc = oc0 + (lane & 15);
  const int hi = lane >> 4;
  const size_t rowbase = ((size_t)b * kC + oc) * kN + n0 + 8 * hi;
#pragma unroll
  for (int r = 0; r < 8; ++r)
    out[rowbase + r] = x[rowbase + r] + gm * acc[r];
}

// ---------------------------------------------------------------------------
extern "C" void kernel_launch(void* const* d_in, const int* in_sizes, int n_in,
                              void* d_out, int out_size, void* d_ws, size_t ws_size,
                              hipStream_t stream) {
  (void)in_sizes; (void)n_in; (void)out_size; (void)ws_size;
  const float* x   = (const float*)d_in[0];
  const float* wt  = (const float*)d_in[1];
  const float* wp  = (const float*)d_in[2];
  const float* wg  = (const float*)d_in[3];
  const float* wo  = (const float*)d_in[4];
  const float* pos = (const float*)d_in[5];
  const float* gam = (const float*)d_in[6];
  float* out = (float*)d_out;

  char* ws = (char*)d_ws;
  size_t off = 0;
  auto alloc = [&](size_t bytes) -> char* {
    char* p = ws + off;
    off += (bytes + 255) & ~(size_t)255;
    return p;
  };
  bf16* Wcat     = (bf16*)alloc((size_t)kCcat * kC * 2);        //  96 KB
  bf16* WoT      = (bf16*)alloc((size_t)kCg * kC * 2);          //  64 KB
  bf16* theta    = (bf16*)alloc((size_t)kB * kCq * kN * 2);     //   2 MB
  bf16* projfull = (bf16*)alloc((size_t)kB * kCpg * kN * 2);    //  10 MB
  bf16* phiT     = (bf16*)alloc((size_t)kB * kM * kCq * 2);     // 0.5 MB
  bf16* gP       = (bf16*)alloc((size_t)kB * kCg * kM * 2);     //   2 MB
  bf16* agT      = (bf16*)alloc((size_t)kB * kN * kCg * 2);     //   8 MB

  prep_weights<<<64, 256, 0, stream>>>(wt, wp, wg, wo, Wcat, WoT);

  // 8 batches * 12 o-tiles * 256 n-tiles, 4 waves/block
  proj_kernel<<<kB * (kCcat / 16) * (kN / 16) / 4, 128, 0, stream>>>(
      x, pos, Wcat, theta, projfull);

  pool_kernel<<<(kB * kCpg * kM + 255) / 256, 256, 0, stream>>>(projfull, phiT, gP);

  // 8 batches * 256 query blocks, 4 waves/block
  attn_kernel<<<kB * (kN / 16) / 4, 128, 0, stream>>>(theta, phiT, gP, agT);

  // 8 batches * 16 oc-tiles * 256 n-tiles, 4 waves/block
  out_kernel<<<kB * (kC / 16) * (kN / 16) / 4, 128, 0, stream>>>(
      x, WoT, agT, gam, out);
}
